// BRNN_85315230368115
// MI455X (gfx1250) — compile-verified, matrix operand order checked by reading, not run on hardware
//
#include <hip/hip_runtime.h>
#include <hip/hip_bf16.h>
#include <math.h>

// ---------------------------------------------------------------------------
// Problem constants
// ---------------------------------------------------------------------------
#define TSTEPS   512           // scan length (axis 0 of x)
#define NSENT    12            // MAX_SENTENCE ("batch" of the scan)
#define NIN      5400          // INPUT
#define KPAD     5408          // NIN padded to multiple of 32 (169*32)
#define NH       256           // HIDDEN
#define NG       1024          // 4*HIDDEN (gates per direction)
#define NGEMM_M  (TSTEPS*NSENT)   // 6144
#define NGEMM_N  (2*NG)           // 2048 (both directions fused)

typedef __attribute__((ext_vector_type(16))) __bf16 v16bf;
typedef __attribute__((ext_vector_type(8)))  __bf16 v8bf;
typedef __attribute__((ext_vector_type(8)))  float  v8f;

__device__ __forceinline__ __bf16 f2bf(float f) {
    unsigned u = __builtin_bit_cast(unsigned, f);
    unsigned r = u + 0x7FFFu + ((u >> 16) & 1u);   // round-to-nearest-even
    unsigned short h = (unsigned short)(r >> 16);
    return __builtin_bit_cast(__bf16, h);
}

__device__ __forceinline__ float sigf(float x) { return 1.f / (1.f + __expf(-x)); }

// Load a 16x32 bf16 WMMA fragment (A, or B from K-major [N][K] storage).
// ISA 7.12.2: lane<16 -> row lane, K {0..7,16..23}; lane>=16 -> row lane-16,
// K {8..15,24..31}.  rowbase points at element [row][kbase].
__device__ __forceinline__ v16bf load_frag(const __bf16* rowbase, int lane) {
    int c0 = (lane >> 4) * 8;                               // 0 or 8
    v8bf lo = *(const v8bf*)(rowbase + c0);
    v8bf hi = *(const v8bf*)(rowbase + c0 + 16);
    return __builtin_shufflevector(lo, hi, 0, 1, 2, 3, 4, 5, 6, 7,
                                           8, 9, 10, 11, 12, 13, 14, 15);
}

// Async (memory -> LDS) 16B copy, tracked by ASYNCcnt.  vdst = LDS byte addr.
__device__ __forceinline__ void async_copy_b128(const void* gaddr, void* lds) {
    unsigned ldsoff = (unsigned)(size_t)lds;                 // LDS aperture low 32b
    unsigned long long ga = (unsigned long long)gaddr;
    asm volatile("global_load_async_to_lds_b128 %0, %1, off"
                 :: "v"(ldsoff), "v"(ga) : "memory");
}
__device__ __forceinline__ void wait_asynccnt0() {
    asm volatile("s_wait_asynccnt 0x0" ::: "memory");
}

// ---------------------------------------------------------------------------
// Conversion kernels (fp32 -> bf16, zero-padded K)
// ---------------------------------------------------------------------------
__global__ void k_conv_x(const float* __restrict__ x, __bf16* __restrict__ xb) {
    size_t i = (size_t)blockIdx.x * blockDim.x + threadIdx.x;
    const size_t total = (size_t)NGEMM_M * KPAD;
    if (i >= total) return;
    int m = (int)(i / KPAD), k = (int)(i % KPAD);
    float v = (k < NIN) ? x[(size_t)m * NIN + k] : 0.f;
    xb[i] = f2bf(v);
}

__global__ void k_conv_wih(const float* __restrict__ Wf, const float* __restrict__ Wb,
                           __bf16* __restrict__ wihb) {
    size_t i = (size_t)blockIdx.x * blockDim.x + threadIdx.x;
    const size_t total = (size_t)NGEMM_N * KPAD;
    if (i >= total) return;
    int n = (int)(i / KPAD), k = (int)(i % KPAD);
    float v = 0.f;
    if (k < NIN)
        v = (n < NG) ? Wf[(size_t)n * NIN + k] : Wb[(size_t)(n - NG) * NIN + k];
    wihb[i] = f2bf(v);
}

__global__ void k_conv_whh(const float* __restrict__ Wf, const float* __restrict__ Wb,
                           __bf16* __restrict__ whhb) {
    size_t i = (size_t)blockIdx.x * blockDim.x + threadIdx.x;
    const size_t total = (size_t)2 * NG * NH;
    if (i >= total) return;
    int dir = (int)(i / (NG * NH));
    int rem = (int)(i % (NG * NH));
    whhb[i] = f2bf(dir ? Wb[rem] : Wf[rem]);
}

__global__ void k_conv_bias(const float* bihf, const float* bhhf,
                            const float* bihb, const float* bhhb,
                            float* __restrict__ biasc) {
    int n = blockIdx.x * blockDim.x + threadIdx.x;
    if (n >= NGEMM_N) return;
    biasc[n] = (n < NG) ? (bihf[n] + bhhf[n]) : (bihb[n - NG] + bhhb[n - NG]);
}

// ---------------------------------------------------------------------------
// Input projection GEMM: G[6144][2048] = Xbf[6144][5408] * Wihbf[2048][5408]^T
// + bias.  Block tile 64Mx64N, 4 waves, each wave owns 16 M-rows.  B tile is
// double-buffered in LDS and staged with GLOBAL_LOAD_ASYNC_TO_LDS_B128 so
// staging of tile k+1 overlaps WMMA on tile k (no VGPR round-trip, no
// per-chunk s_wait_loadcnt).  A is loaded direct (L2-resident).
// ---------------------------------------------------------------------------
__device__ __forceinline__ void stage_b_async(const __bf16* __restrict__ W,
                                              __bf16 (*Bs)[32], int n0, int k,
                                              int tid) {
    // 64 rows x 32 bf16 = 4KB = 256 chunks of 16B; 2 chunks per thread
#pragma unroll
    for (int c = tid; c < 64 * 4; c += 128) {
        int r = c >> 2, cc = (c & 3) * 8;
        async_copy_b128(W + (size_t)(n0 + r) * KPAD + k + cc, &Bs[r][cc]);
    }
}

__global__ void __launch_bounds__(128, 2)
k_gemm_in(const __bf16* __restrict__ X, const __bf16* __restrict__ W,
          const float* __restrict__ biasc, float* __restrict__ G) {
    __shared__ __bf16 Bs[2][64][32];
    const int tid  = threadIdx.x;
    const int w    = tid >> 5;
    const int lane = tid & 31;
    const int m0   = blockIdx.x * 64 + w * 16;
    const int n0   = blockIdx.y * 64;

    v8f acc[4] = {};
    const __bf16* Arow = X + (size_t)(m0 + (lane & 15)) * KPAD;

    // prologue: stage tile 0 asynchronously
    stage_b_async(W, Bs[0], n0, 0, tid);
    wait_asynccnt0();
    __syncthreads();

    for (int k = 0; k < KPAD; k += 32) {
        const int cur = (k >> 5) & 1;
        if (k + 32 < KPAD)                       // overlap: stage next tile
            stage_b_async(W, Bs[cur ^ 1], n0, k + 32, tid);
        __builtin_prefetch(Arow + k + 32, 0, 1); // global_prefetch_b8 (A stream)

        v16bf a = load_frag(Arow + k, lane);
#pragma unroll
        for (int nt = 0; nt < 4; ++nt) {
            v16bf b = load_frag(&Bs[cur][nt * 16 + (lane & 15)][0], lane);
            acc[nt] = __builtin_amdgcn_wmma_f32_16x16x32_bf16(
                false, a, false, b, (short)0, acc[nt], false, false);
        }
        wait_asynccnt0();                        // own async stores done
        __syncthreads();                         // everyone's done -> flip buffers
    }

    // epilogue: C layout -> (row = r + 8*(lane>=16), col = lane%16)
    const int col  = lane & 15;
    const int rsel = (lane >> 4) * 8;
#pragma unroll
    for (int nt = 0; nt < 4; ++nt) {
        int n = n0 + nt * 16 + col;
        float bias = biasc[n];
#pragma unroll
        for (int r = 0; r < 8; ++r) {
            int row = m0 + r + rsel;
            G[(size_t)row * NGEMM_N + n] = acc[nt][r] + bias;
        }
    }
}

// ---------------------------------------------------------------------------
// Recurrent scan: 1 persistent workgroup per direction (grid=2, 256 thr = 8
// waves).  Per step: gates(12x1024) = h(12x256, bf16 LDS) @ Whh^T via WMMA,
// + precomputed G, activations, new h -> LDS + H output.
// ---------------------------------------------------------------------------
__global__ void __launch_bounds__(256, 1)
k_lstm_scan(const float* __restrict__ Gc,     // [6144][2048]
            const __bf16* __restrict__ Whh,   // [2][1024][256] (K-major)
            float* __restrict__ H)            // [512][12][512]
{
    __shared__ __bf16 hbuf[16][NH];           // h state (rows 12..15 stay 0)
    __shared__ float  gbuf[NSENT][NG];        // recurrent part of gates

    const int dir  = blockIdx.x;
    const int tid  = threadIdx.x;
    const int w    = tid >> 5;
    const int lane = tid & 31;

    for (int i = tid; i < 16 * NH; i += 256)
        ((__bf16*)hbuf)[i] = __builtin_bit_cast(__bf16, (unsigned short)0);

    float c[NSENT];
#pragma unroll
    for (int s = 0; s < NSENT; ++s) c[s] = 0.f;
    __syncthreads();

    const __bf16* Wd = Whh + (size_t)dir * NG * NH;

    for (int it = 0; it < TSTEPS; ++it) {
        const int t = dir ? (TSTEPS - 1 - it) : it;

        // ---- h @ Whh^T : each wave owns 8 n-tiles of 16 ----
        v8f acc[8] = {};
        for (int k = 0; k < NH; k += 32) {
            v16bf a = load_frag(&hbuf[lane & 15][k], lane);
#pragma unroll
            for (int nt = 0; nt < 8; ++nt) {
                int n = (w * 8 + nt) * 16 + (lane & 15);
                v16bf b = load_frag(Wd + (size_t)n * NH + k, lane);
                acc[nt] = __builtin_amdgcn_wmma_f32_16x16x32_bf16(
                    false, a, false, b, (short)0, acc[nt], false, false);
            }
        }
        const int col  = lane & 15;
        const int rsel = (lane >> 4) * 8;
#pragma unroll
        for (int nt = 0; nt < 8; ++nt) {
            int n = (w * 8 + nt) * 16 + col;
#pragma unroll
            for (int r = 0; r < 8; ++r) {
                int row = r + rsel;
                if (row < NSENT) gbuf[row][n] = acc[nt][r];
            }
        }
        __syncthreads();

        // ---- activations: thread j handles hidden unit j for all sentences
        const float* Gt = Gc + (size_t)t * NSENT * NGEMM_N + (size_t)dir * NG;
        const int j = tid;
#pragma unroll
        for (int s = 0; s < NSENT; ++s) {
            const float* Gs = Gt + (size_t)s * NGEMM_N;
            float ii = gbuf[s][j]          + Gs[j];
            float ff = gbuf[s][NH + j]     + Gs[NH + j];
            float gg = gbuf[s][2 * NH + j] + Gs[2 * NH + j];
            float oo = gbuf[s][3 * NH + j] + Gs[3 * NH + j];
            float cs = sigf(ff) * c[s] + sigf(ii) * tanhf(gg);
            c[s] = cs;
            float h = sigf(oo) * tanhf(cs);
            hbuf[s][j] = f2bf(h);
            H[((size_t)t * NSENT + s) * (2 * NH) + dir * NH + j] = h;
        }
        __syncthreads();
    }
}

// ---------------------------------------------------------------------------
// Per-document epilogue (one block / doc, 128 threads)
// ---------------------------------------------------------------------------
__global__ void __launch_bounds__(128, 2)
k_post(const float* __restrict__ H,     // [512][12][512]
       const float* __restrict__ encW,  // [50][512]
       const float* __restrict__ encb,  // [50]
       const float* __restrict__ wfcW,  // [100]
       const float* __restrict__ wfcb,  // [1]
       const float* __restrict__ fcW,   // [2][12]
       const float* __restrict__ fcb,   // [2]
       float* __restrict__ out)         // [512][2]
{
    __shared__ float embA[NSENT][512];
    __shared__ float embB[NSENT][512];
    __shared__ float enc[NSENT][50];
    __shared__ float av[NSENT], bv[NSENT], nrm[NSENT], simv[NSENT], rsc[NSENT];
    __shared__ float Wm[NSENT][NSENT];
    __shared__ int   core_s;

    const int d = blockIdx.x, tid = threadIdx.x;
    const float* Hd = H + (size_t)d * NSENT * 512;

    for (int i = tid; i < NSENT * 512; i += 128) ((float*)embA)[i] = Hd[i];
    __syncthreads();

    // enc = leaky_relu(H @ encW^T + encb)
    for (int p = tid; p < NSENT * 50; p += 128) {
        int s = p / 50, e = p - s * 50;
        float acc = encb[e];
        const float* we = encW + (size_t)e * 512;
        for (int k = 0; k < 512; ++k) acc = fmaf(embA[s][k], we[k], acc);
        enc[s][e] = acc > 0.f ? acc : 0.01f * acc;
    }
    if (tid < NSENT) {
        float acc = 0.f;
        for (int k = 0; k < 512; ++k) acc = fmaf(embA[tid][k], embA[tid][k], acc);
        nrm[tid] = sqrtf(acc) + 1e-12f;
    }
    __syncthreads();

    if (tid < 2 * NSENT) {
        int s = tid % NSENT, half = (tid / NSENT) * 50;
        float acc = 0.f;
        for (int e = 0; e < 50; ++e) acc = fmaf(enc[s][e], wfcW[half + e], acc);
        if (half) bv[s] = acc; else av[s] = acc;
    }
    __syncthreads();

    if (tid == 0) {   // rowsum argmax -> core
        float best = -3.4e38f; int bi = 0;
        for (int i = 0; i < NSENT; ++i) {
            float rs = 0.f;
            for (int j = 0; j < NSENT; ++j)
                if (j != i) rs += av[i] + bv[j] + wfcb[0];
            if (rs > best) { best = rs; bi = i; }
        }
        core_s = bi;
    }
    __syncthreads();
    const int core = core_s;

    if (tid < NSENT) {
        float acc = 0.f;
        for (int k = 0; k < 512; ++k) acc = fmaf(embA[tid][k], embA[core][k], acc);
        simv[tid] = acc / (nrm[tid] * nrm[core]);
    }
    __syncthreads();

    if (tid == 0) {   // stable descending ranks -> rank_score
        for (int s = 0; s < NSENT; ++s) {
            int rank = 0;
            for (int s2 = 0; s2 < NSENT; ++s2)
                if (simv[s2] > simv[s] || (simv[s2] == simv[s] && s2 < s)) rank++;
            rsc[s] = 1.f - (float)rank / (float)NSENT;
        }
    }
    __syncthreads();

    if (tid < NSENT * NSENT) {
        int i = tid / NSENT, j = tid - i * NSENT;
        Wm[i][j] = (i == j) ? 0.f : (av[i] + bv[j] + wfcb[0]) * rsc[i];
    }
    __syncthreads();

    // emb <- emb + Wm @ emb   (twice)
    for (int p = tid; p < NSENT * 512; p += 128) {
        int i = p >> 9, h = p & 511;
        float acc = embA[i][h];
#pragma unroll
        for (int j = 0; j < NSENT; ++j) acc = fmaf(Wm[i][j], embA[j][h], acc);
        embB[i][h] = acc;
    }
    __syncthreads();
    for (int p = tid; p < NSENT * 512; p += 128) {
        int i = p >> 9, h = p & 511;
        float acc = embB[i][h];
#pragma unroll
        for (int j = 0; j < NSENT; ++j) acc = fmaf(Wm[i][j], embB[j][h], acc);
        embA[i][h] = acc;
    }
    __syncthreads();

    if (tid < NSENT) {   // doc = mean over hidden (reuse av)
        float acc = 0.f;
        for (int k = 0; k < 512; ++k) acc += embA[tid][k];
        av[tid] = acc / 512.f;
    }
    __syncthreads();
    if (tid < 2) {
        float acc = fcb[tid];
        for (int s = 0; s < NSENT; ++s) acc = fmaf(av[s], fcW[tid * NSENT + s], acc);
        out[d * 2 + tid] = acc;
    }
}

// ---------------------------------------------------------------------------
// Host launcher
// ---------------------------------------------------------------------------
extern "C" void kernel_launch(void* const* d_in, const int* in_sizes, int n_in,
                              void* d_out, int out_size, void* d_ws, size_t ws_size,
                              hipStream_t stream) {
    (void)in_sizes; (void)n_in; (void)out_size; (void)ws_size;

    const float* x     = (const float*)d_in[0];
    const float* Wih_f = (const float*)d_in[1];
    const float* Whh_f = (const float*)d_in[2];
    const float* bih_f = (const float*)d_in[3];
    const float* bhh_f = (const float*)d_in[4];
    const float* Wih_b = (const float*)d_in[5];
    const float* Whh_b = (const float*)d_in[6];
    const float* bih_b = (const float*)d_in[7];
    const float* bhh_b = (const float*)d_in[8];
    const float* encW  = (const float*)d_in[9];
    const float* encb  = (const float*)d_in[10];
    const float* wfcW  = (const float*)d_in[11];
    const float* wfcb  = (const float*)d_in[12];
    const float* fcW   = (const float*)d_in[13];
    const float* fcb   = (const float*)d_in[14];
    float* out = (float*)d_out;

    // workspace carve-up (256B aligned)
    size_t off = 0;
    auto carve = [&](size_t bytes) {
        void* p = (char*)d_ws + off;
        off += (bytes + 255) & ~(size_t)255;
        return p;
    };
    __bf16* xb    = (__bf16*)carve((size_t)NGEMM_M * KPAD * 2);   // 66.5 MB
    __bf16* wihb  = (__bf16*)carve((size_t)NGEMM_N * KPAD * 2);   // 22.2 MB
    __bf16* whhb  = (__bf16*)carve((size_t)2 * NG * NH * 2);      //  1.0 MB
    float*  biasc = (float*) carve((size_t)NGEMM_N * 4);
    float*  Gc    = (float*) carve((size_t)NGEMM_M * NGEMM_N * 4);// 50.3 MB
    float*  Hbuf  = (float*) carve((size_t)TSTEPS * NSENT * 2 * NH * 4); // 12.6 MB

    // 1) convert inputs/weights to bf16 (K zero-padded)
    {
        size_t n = (size_t)NGEMM_M * KPAD;
        k_conv_x<<<dim3((unsigned)((n + 255) / 256)), dim3(256), 0, stream>>>(x, xb);
    }
    {
        size_t n = (size_t)NGEMM_N * KPAD;
        k_conv_wih<<<dim3((unsigned)((n + 255) / 256)), dim3(256), 0, stream>>>(Wih_f, Wih_b, wihb);
    }
    {
        size_t n = (size_t)2 * NG * NH;
        k_conv_whh<<<dim3((unsigned)((n + 255) / 256)), dim3(256), 0, stream>>>(Whh_f, Whh_b, whhb);
    }
    k_conv_bias<<<dim3(8), dim3(256), 0, stream>>>(bih_f, bhh_f, bih_b, bhh_b, biasc);

    // 2) fused input-projection GEMM for both directions (WMMA bf16,
    //    async-to-LDS double-buffered B staging)
    k_gemm_in<<<dim3(NGEMM_M / 64, NGEMM_N / 64), dim3(128), 0, stream>>>(xb, wihb, biasc, Gc);

    // 3) bidirectional recurrent scan (one persistent WG per direction)
    k_lstm_scan<<<dim3(2), dim3(256), 0, stream>>>(Gc, whhb, Hbuf);

    // 4) per-document epilogue
    k_post<<<dim3(TSTEPS), dim3(128), 0, stream>>>(Hbuf, encW, encb, wfcW, wfcb, fcW, fcb, out);
}
